// GeoMamba_59227599012344
// MI455X (gfx1250) — matrix operand order repaired and checked by reading, not compiled, not run on hardware
//
#include <hip/hip_runtime.h>
#include <hip/hip_bf16.h>
#include <math.h>

typedef __attribute__((ext_vector_type(16))) _Float16 v16h;
typedef __attribute__((ext_vector_type(8)))  _Float16 v8h;
typedef __attribute__((ext_vector_type(4)))  _Float16 v4h;
typedef __attribute__((ext_vector_type(8)))  float    v8f;

#define BB    8
#define LL    39
#define DDIM  256
#define DSQ   65536      // D*D
#define MMM   252        // D-K+1
#define MMSQ  63504      // M*M (= 3969*16)
#define KCONV 975        // 39*5*5
#define PADK  992        // KCONV padded to 32
#define PADF  63520      // MMSQ padded to 32

// WMMA fragment k order used by this file (A and B alike), lane = m + 16h:
//   A frag element e -> K = e + 16h         == one contiguous 16-elem run at 16h
//   B frag element e -> K = e + 16h         == one contiguous 16-elem run at 16h
union HPair { v16h v; struct { v8h lo, hi; } p; };

__device__ __forceinline__ v16h ld_a16(const _Float16* base, int h) {
    HPair u;
    u.p.lo = *(const v8h*)(base + 16 * h);
    u.p.hi = *(const v8h*)(base + 16 * h + 8);
    return u.v;
}
__device__ __forceinline__ v16h ld_b16(const _Float16* base, int h) {
    HPair u;
    u.p.lo = *(const v8h*)(base + 16 * h);
    u.p.hi = *(const v8h*)(base + 16 * h + 8);
    return u.v;
}

// ---------------------------------------------------------------------------
// Elementwise prep kernels (one-time f32 -> f16 conversions, zero-padded K)
// ---------------------------------------------------------------------------
__global__ __launch_bounds__(256) void k_zero(float* p, int n) {
    int i = blockIdx.x * blockDim.x + threadIdx.x;
    if (i < n) p[i] = 0.0f;
}

__global__ __launch_bounds__(256) void k_cvt4(const float* __restrict__ src,
                                              _Float16* __restrict__ dst, int n4) {
    int i = blockIdx.x * blockDim.x + threadIdx.x;
    if (i >= n4) return;
    float4 v = ((const float4*)src)[i];
    v4h o = { (_Float16)v.x, (_Float16)v.y, (_Float16)v.z, (_Float16)v.w };
    ((v4h*)dst)[i] = o;
}

__global__ __launch_bounds__(256) void k_cvt_cw(const float* __restrict__ cw,
                                                _Float16* __restrict__ cwH) {
    int idx = blockIdx.x * blockDim.x + threadIdx.x;
    if (idx >= LL * PADK) return;
    int o = idx / PADK, k = idx % PADK;
    cwH[idx] = (k < KCONV) ? (_Float16)cw[o * KCONV + k] : (_Float16)0.0f;
}

__global__ __launch_bounds__(256) void k_cvt_ow(const float* __restrict__ ow,
                                                _Float16* __restrict__ owH) {
    int idx = blockIdx.x * blockDim.x + threadIdx.x;   // 256*63520 elements
    int o = idx / PADF, k = idx % PADF;
    owH[idx] = (k < MMSQ) ? (_Float16)ow[(size_t)o * MMSQ + k] : (_Float16)0.0f;
}

// zH[row][k] = tril-masked, 1/norm-scaled conv activation, f16, zero-padded
__global__ __launch_bounds__(256) void k_zh(const float* __restrict__ y,
                                            const float* __restrict__ invn,
                                            _Float16* __restrict__ zH) {
    long idx = (long)blockIdx.x * blockDim.x + threadIdx.x;  // 312*63520
    if (idx >= (long)BB * LL * PADF) return;
    int row = (int)(idx / PADF), k = (int)(idx % PADF);
    _Float16 v = (_Float16)0.0f;
    if (k < MMSQ) {
        int r = k / MMM, c = k - r * MMM;
        if (c <= r) v = (_Float16)(y[(size_t)row * MMSQ + k] * invn[row]);
    }
    zH[idx] = v;
}

// ---------------------------------------------------------------------------
// Gram matrix with DOUBLE-BUFFERED async-LDS staging.
// Grid: 8 batches x 32 K-slices; block = 9 waves (288 thr) = 3x3 output tiles.
// Pipeline per step s: issue async copy of chunk s+1 into buf[(s+1)&1],
// s_wait_asynccnt 1 (chunk s's copy, issued last step, is proven complete:
// ASYNCcnt retires in issue order), barrier, consume buf[s&1] via ds_load_b128
// into WMMA, barrier (protects buf[s&1] readers before its overwrite).
// Partial sums accumulate with global_atomic_add_f32 into pre-zeroed scores.
// ---------------------------------------------------------------------------
__global__ __launch_bounds__(288) void k_gram(const _Float16* __restrict__ hsH,
                                              float* __restrict__ scores) {
    __shared__ __align__(16) _Float16 tile[2][48 * 32];
    int tid = threadIdx.x;
    int lane = tid & 31;
    int wv = tid >> 5;                        // 0..8
    int m = lane & 15, h = lane >> 4;
    int b = blockIdx.x >> 5;
    int slice = blockIdx.x & 31;              // K slice: 2048 elements each
    const _Float16* Hb = hsH + (size_t)b * LL * DSQ;

    // zero pad rows 39..47 of both buffers (copies touch only rows 0..38)
    for (int i = tid; i < 2 * 9 * 32; i += 288) {
        int t = i / (9 * 32);
        int rr = (i >> 5) % 9;
        tile[t][(39 + rr) * 32 + (i & 31)] = (_Float16)0.0f;
    }
    __syncthreads();

    int ti = wv / 3, tj = wv % 3;
    int srow = tid >> 2, sseg = tid & 3;      // 4 x 16B segments per row
    bool do_copy = (srow < LL);
    unsigned lds0 = (unsigned)(size_t)(&tile[0][srow * 32 + sseg * 8]);
    unsigned lds1 = (unsigned)(size_t)(&tile[1][srow * 32 + sseg * 8]);
    const _Float16* gsrc0 = Hb + (size_t)srow * DSQ + sseg * 8 + slice * 2048;

    if (do_copy) {  // prologue: chunk 0 -> buf0
        asm volatile("global_load_async_to_lds_b128 %0, %1, off"
                     :: "v"(lds0), "v"(gsrc0) : "memory");
    }

    v8f c = {};
    for (int s = 0; s < 64; ++s) {
        int nxt = s + 1;
        if (nxt < 64) {
            if (do_copy) {
                const _Float16* g = gsrc0 + (size_t)nxt * 32;
                unsigned dst = (nxt & 1) ? lds1 : lds0;
                asm volatile("global_load_async_to_lds_b128 %0, %1, off"
                             :: "v"(dst), "v"(g) : "memory");
            }
            asm volatile("s_wait_asynccnt 1" ::: "memory");   // chunk s landed
        } else {
            asm volatile("s_wait_asynccnt 0" ::: "memory");   // last chunk landed
        }
        __syncthreads();
        const _Float16* tb = tile[s & 1];
        v16h a  = ld_a16(tb + (ti * 16 + m) * 32, h);
        v16h bb = ld_b16(tb + (tj * 16 + m) * 32, h);
        c = __builtin_amdgcn_wmma_f32_16x16x32_f16(false, a, false, bb,
                                                   (short)0, c, false, false);
        __syncthreads();
    }
#pragma unroll
    for (int r = 0; r < 8; ++r) {
        int i = ti * 16 + r + 8 * h;
        int j = tj * 16 + m;
        if (i < LL && j < LL)
            atomicAdd(&scores[((size_t)b * LL + i) * LL + j], c[r] * (1.0f / 256.0f));
    }
}

// ---------------------------------------------------------------------------
// Softmax over 312 rows of 39. One thread per row.
// ---------------------------------------------------------------------------
__global__ __launch_bounds__(256) void k_softmax(const float* __restrict__ scores,
                                                 float* __restrict__ attn) {
    int idx = blockIdx.x * blockDim.x + threadIdx.x;
    if (idx >= BB * LL) return;
    const float* s = scores + (size_t)idx * LL;
    float mx = -1e30f;
    for (int j = 0; j < LL; ++j) mx = fmaxf(mx, s[j]);
    float sum = 0.f;
    for (int j = 0; j < LL; ++j) sum += __expf(s[j] - mx);
    float inv = 1.f / sum;
    float* a = attn + (size_t)idx * LL;
    for (int j = 0; j < LL; ++j) a[j] = __expf(s[j] - mx) * inv;
}

// ---------------------------------------------------------------------------
// xH[b] = attention[b] (39x39) @ hsH[b] (39x65536), K pad 39->64, f16 out.
// ---------------------------------------------------------------------------
__global__ __launch_bounds__(128) void k_attn_apply(const float* __restrict__ attn,
                                                    const _Float16* __restrict__ hsH,
                                                    _Float16* __restrict__ xH) {
    int lane = threadIdx.x & 31;
    int wave = threadIdx.x >> 5;
    int m = lane & 15, h = lane >> 4;
    int w = blockIdx.x * 4 + wave;
    int b  = w / (3 * 4096);
    int r  = w % (3 * 4096);
    int ti = r / 4096;
    int tn = r % 4096;
    int rowi = ti * 16 + m; if (rowi >= LL) rowi = LL - 1;
    const float* A      = attn + ((size_t)b * LL + rowi) * LL;
    const _Float16* Hb  = hsH + (size_t)b * LL * DSQ;
    int ncol = tn * 16 + m;

    v8f c = {};
#pragma unroll
    for (int k0 = 0; k0 < 64; k0 += 32) {
        v16h a, bf;
#pragma unroll
        for (int e = 0; e < 16; ++e) {
            int ka = k0 + 16 * h + e;
            a[e] = (ka < LL) ? (_Float16)A[ka] : (_Float16)0.0f;
            int kb = k0 + 16 * h + e;
            bf[e] = (kb < LL) ? Hb[(size_t)kb * DSQ + ncol] : (_Float16)0.0f;
        }
        c = __builtin_amdgcn_wmma_f32_16x16x32_f16(false, a, false, bf,
                                                   (short)0, c, false, false);
    }
    _Float16* X = xH + (size_t)b * LL * DSQ;
#pragma unroll
    for (int rr = 0; rr < 8; ++rr) {
        int i = ti * 16 + rr + 8 * h;
        if (i < LL) X[(size_t)i * DSQ + ncol] = (_Float16)c[rr];
    }
}

// ---------------------------------------------------------------------------
// Implicit-GEMM conv + bias + exp. A = padded f16 weights (pure b128 loads),
// B = f16 activation gather via precomputed LDS offset table, read back with
// ds_load_b128 (int4). K = 992 (zero-padded), 31 WMMA steps per tile.
// ---------------------------------------------------------------------------
__global__ __launch_bounds__(128) void k_conv(const _Float16* __restrict__ xH,
                                              const _Float16* __restrict__ cwH,
                                              const float* __restrict__ cb,
                                              float* __restrict__ y) {
    __shared__ __align__(16) int offt[PADK];
    for (int k = threadIdx.x; k < PADK; k += 128) {
        int of = -1;
        if (k < KCONV) {
            int ic = k / 25, rem = k - ic * 25;
            int ry = rem / 5, rx = rem - ry * 5;
            of = ic * DSQ + ry * DDIM + rx;
        }
        offt[k] = of;
    }
    __syncthreads();

    int lane = threadIdx.x & 31;
    int wave = threadIdx.x >> 5;
    int m = lane & 15, h = lane >> 4;
    int w = blockIdx.x * 4 + wave;
    int b  = w / (3 * 3969);
    int r  = w % (3 * 3969);
    int to = r / 3969;
    int tp = r % 3969;
    int o  = to * 16 + m; int oc = (o >= LL) ? LL - 1 : o;
    const _Float16* W = cwH + (size_t)oc * PADK;
    int p  = tp * 16 + m;
    int py = p / MMM, px = p % MMM;
    const _Float16* P = xH + (size_t)b * LL * DSQ + py * DDIM + px;

    v8f c = {};
    for (int k0 = 0; k0 < PADK; k0 += 32) {
        v16h a = ld_a16(W + k0, h);
        v16h bf;
        const int4* ot = (const int4*)&offt[k0 + 16 * h];   // 16B-aligned LDS
#pragma unroll
        for (int q = 0; q < 4; ++q) {
            int4 ov = ot[q];
            bf[q * 4 + 0] = (ov.x >= 0) ? P[ov.x] : (_Float16)0.0f;
            bf[q * 4 + 1] = (ov.y >= 0) ? P[ov.y] : (_Float16)0.0f;
            bf[q * 4 + 2] = (ov.z >= 0) ? P[ov.z] : (_Float16)0.0f;
            bf[q * 4 + 3] = (ov.w >= 0) ? P[ov.w] : (_Float16)0.0f;
        }
        c = __builtin_amdgcn_wmma_f32_16x16x32_f16(false, a, false, bf,
                                                   (short)0, c, false, false);
    }
#pragma unroll
    for (int rr = 0; rr < 8; ++rr) {
        int oo = to * 16 + rr + 8 * h;
        if (oo < LL)
            y[((size_t)b * LL + oo) * MMSQ + p] = __expf(c[rr] + cb[oo]);
    }
}

// ---------------------------------------------------------------------------
// Frobenius 1/norm per (b, channel).
// ---------------------------------------------------------------------------
__global__ __launch_bounds__(256) void k_norm(const float* __restrict__ y,
                                              float* __restrict__ invn) {
    __shared__ float red[256];
    int bl = blockIdx.x;
    const float* Y = y + (size_t)bl * MMSQ;
    float s = 0.f;
    for (int i = threadIdx.x; i < MMSQ; i += 256) { float v = Y[i]; s += v * v; }
    red[threadIdx.x] = s;
    __syncthreads();
    for (int off = 128; off > 0; off >>= 1) {
        if (threadIdx.x < off) red[threadIdx.x] += red[threadIdx.x + off];
        __syncthreads();
    }
    if (threadIdx.x == 0) invn[bl] = rsqrtf(red[0]);
}

// ---------------------------------------------------------------------------
// out[b,l,:] = zH[b,l,:] @ owH^T. Fully padded K (1985 x 32), pure b128
// loads + WMMA + prefetch-ahead of the streamed activation row.
// ---------------------------------------------------------------------------
__global__ __launch_bounds__(128) void k_proj(const _Float16* __restrict__ zH,
                                              const _Float16* __restrict__ owH,
                                              float* __restrict__ out) {
    int lane = threadIdx.x & 31;
    int wave = threadIdx.x >> 5;
    int m = lane & 15, h = lane >> 4;
    int w = blockIdx.x * 4 + wave;   // 0..383
    int b  = w / 48;
    int r  = w % 48;
    int tl = r / 16;
    int tn = r % 16;
    int l  = tl * 16 + m; int lc = (l >= LL) ? LL - 1 : l;
    const _Float16* Y  = zH + ((size_t)b * LL + lc) * PADF;
    int o = tn * 16 + m;
    const _Float16* Wo = owH + (size_t)o * PADF;

    v8f c = {};
    for (int k0 = 0; k0 < PADF; k0 += 32) {
        __builtin_prefetch(Y + k0 + 1024, 0, 1);   // global_prefetch_b8 ~2KB ahead
        v16h a  = ld_a16(Y + k0, h);
        v16h bf = ld_b16(Wo + k0, h);
        c = __builtin_amdgcn_wmma_f32_16x16x32_f16(false, a, false, bf,
                                                   (short)0, c, false, false);
    }
#pragma unroll
    for (int rr = 0; rr < 8; ++rr) {
        int li = tl * 16 + rr + 8 * h;
        if (li < LL)
            out[((size_t)b * LL + li) * DDIM + tn * 16 + m] = c[rr];
    }
}

// ---------------------------------------------------------------------------
extern "C" void kernel_launch(void* const* d_in, const int* in_sizes, int n_in,
                              void* d_out, int out_size, void* d_ws, size_t ws_size,
                              hipStream_t stream) {
    const float* hs = (const float*)d_in[0];   // (8,39,256,256)
    const float* cw = (const float*)d_in[1];   // (512,39,5,5) -> first 39 rows used
    const float* cb = (const float*)d_in[2];   // (512,)
    const float* ow = (const float*)d_in[3];   // (256,63504)

    float* out  = (float*)d_out;                    // 79872
    float* attn = out + (size_t)BB * LL * DDIM;     // 12168

    // workspace carve-up (all 16B aligned)
    char* base = (char*)d_ws;
    _Float16* hsH = (_Float16*)base;            base += (size_t)BB * LL * DSQ * 2;   // 40.9 MB
    _Float16* xH  = (_Float16*)base;            base += (size_t)BB * LL * DSQ * 2;   // 40.9 MB
    float*    y   = (float*)base;               base += (size_t)BB * LL * MMSQ * 4;  // 79.3 MB
    _Float16* zH  = (_Float16*)base;            base += (size_t)BB * LL * PADF * 2;  // 39.6 MB
    _Float16* owH = (_Float16*)base;            base += (size_t)DDIM * PADF * 2;     // 32.5 MB
    _Float16* cwH = (_Float16*)base;            base += (size_t)LL * PADK * 2;
    float* scores = (float*)base;               base += (size_t)BB * LL * LL * 4;
    float* invn   = (float*)base;

    int n4 = (BB * LL * DSQ) / 4;

    k_zero      <<<48,                  256, 0, stream>>>(scores, BB * LL * LL);
    k_cvt4      <<<n4 / 256,            256, 0, stream>>>(hs, hsH, n4);
    k_cvt_cw    <<<(LL * PADK + 255) / 256, 256, 0, stream>>>(cw, cwH);
    k_cvt_ow    <<<DDIM * PADF / 256,   256, 0, stream>>>(ow, owH);
    k_gram      <<<BB * 32,             288, 0, stream>>>(hsH, scores);
    k_softmax   <<<2,                   256, 0, stream>>>(scores, attn);
    k_attn_apply<<<(BB * 3 * 4096) / 4, 128, 0, stream>>>(attn, hsH, xH);
    k_conv      <<<(BB * 3 * 3969) / 4, 128, 0, stream>>>(xH, cwH, cb, y);
    k_norm      <<<BB * LL,             256, 0, stream>>>(y, invn);
    k_zh        <<<(BB * LL * PADF) / 256, 256, 0, stream>>>(y, invn, zH);
    k_proj      <<<(BB * 3 * 16) / 4,   128, 0, stream>>>(zH, owH, out);
}
